// One2ManyAttention_86320252715444
// MI455X (gfx1250) — matrix operand order repaired and checked by reading, not compiled, not run on hardware
//
#include <hip/hip_runtime.h>

typedef __attribute__((ext_vector_type(16))) _Float16 v16h;
typedef __attribute__((ext_vector_type(8)))  float    v8f;

#define NHEAD    8
#define HEAD_DIM 32
#define H_       32
#define W_       40
#define L_       (H_ * W_)          // 1280
#define FEAT     (NHEAD * HEAD_DIM) // 256

__device__ __forceinline__ void inv3(const float m[9], float o[9]) {
    float a=m[0],b=m[1],c=m[2],d=m[3],e=m[4],f=m[5],g=m[6],h=m[7],i=m[8];
    float A =  (e*i - f*h);
    float B = -(d*i - f*g);
    float C =  (d*h - e*g);
    float det = a*A + b*B + c*C;
    float r = 1.0f / det;
    o[0]=A*r;            o[1]=-(b*i - c*h)*r;  o[2]= (b*f - c*e)*r;
    o[3]=B*r;            o[4]= (a*i - c*g)*r;  o[5]=-(a*f - c*d)*r;
    o[6]=C*r;            o[7]=-(a*h - b*g)*r;  o[8]= (a*e - b*d)*r;
}

__device__ __forceinline__ void mm3(const float* A, const float* B, float* C) {
    #pragma unroll
    for (int i = 0; i < 3; i++)
        #pragma unroll
        for (int j = 0; j < 3; j++)
            C[i*3+j] = A[i*3+0]*B[0*3+j] + A[i*3+1]*B[1*3+j] + A[i*3+2]*B[2*3+j];
}

// One workgroup = ONE wave (32 threads): handles a 16-pixel tile for one head.
// Single-wave workgroups need no LDS/barriers and spread across all WGPs.
// Flash-attention over all 1280 candidates with an on-the-fly epipolar-band
// mask (equivalent to the reference's gather+sort+softmax, since the softmax
// weighted sum is permutation invariant and the candidate count never exceeds
// C_MAX for a width-4 band on a 32x40 grid).
__global__ __launch_bounds__(32)
void One2ManyAttention_kernel(const float* __restrict__ q,
                              const float* __restrict__ k,
                              const float* __restrict__ v,
                              const float* __restrict__ K0,
                              const float* __restrict__ K1,
                              const float* __restrict__ R,
                              const float* __restrict__ t,
                              float* __restrict__ out)
{
    const int lane = threadIdx.x & 31;
    const int h    = blockIdx.y;     // head 0..7
    const int lt   = blockIdx.x;     // pixel tile 0..79
    const int n    = lane & 15;      // column within tile (pixel) / B col
    const int hi   = lane >> 4;      // lane-half per WMMA layouts

    // ---- F = inv(K1)^T * skew(t) * R * inv(K0) (computed redundantly per lane)
    float k0i[9], k1i[9], tA[9], tB[9], F[9];
    {
        float m[9];
        #pragma unroll
        for (int i = 0; i < 9; i++) m[i] = K0[i];
        inv3(m, k0i);
        #pragma unroll
        for (int i = 0; i < 9; i++) m[i] = K1[i];
        inv3(m, k1i);
    }
    const float S[9] = { 0.0f, -t[2],  t[1],
                         t[2],  0.0f, -t[0],
                        -t[1],  t[0],  0.0f };
    mm3(S, R, tA);       // skew(t) * R
    mm3(tA, k0i, tB);    // * inv(K0)
    #pragma unroll
    for (int i = 0; i < 3; i++)
        #pragma unroll
        for (int j = 0; j < 3; j++) {
            float s = 0.0f;
            #pragma unroll
            for (int kk = 0; kk < 3; kk++) s += k1i[kk*3 + i] * tB[kk*3 + j];
            F[i*3 + j] = s;   // inv(K1)^T * tB
        }

    // ---- epipolar line of the pixel this lane owns (tile column n)
    const int   l  = lt * 16 + n;
    const float xl = (float)(l % W_);
    const float yl = (float)(l / W_);
    const float la = F[0]*xl + F[1]*yl + F[2];
    const float lb = F[3]*xl + F[4]*yl + F[5];
    const float lc = F[6]*xl + F[7]*yl + F[8];
    const bool  mode = fabsf(lb) > fabsf(la);
    const float den  = mode ? lb : la;
    const float slp  = (mode ? -la : -lb) / den;  // NaN/Inf if degenerate -> mask false (matches ref)
    const float icp  = -lc / den;

    // ---- B fragment for scores WMMA: B = Q^T (32 dims x 16 pixels)
    v16h bQ;
    {
        const float* qp = q + l * FEAT + h * HEAD_DIM + hi * 16;
        #pragma unroll
        for (int e = 0; e < 16; e++) bQ[e] = (_Float16)qp[e];
    }

    const float scale = 0.17677669529663687f;   // 1/sqrt(32)

    float run_max = -3.0e38f;
    float run_sum = 0.0f;
    v8f acc0 = {};   // out rows (hi*8+r) x dims 0..15
    v8f acc1 = {};   // out rows (hi*8+r) x dims 16..31

    for (int chunk = 0; chunk < L_; chunk += 32) {
        float msc[2][8];
        float localmax = -3.0e38f;

        // two 16-candidate subtiles: transposed scores D[cand][pixel]
        #pragma unroll
        for (int sub = 0; sub < 2; sub++) {
            const int sbase = chunk + sub * 16;
            // A fragment = K tile (16 candidates x 32 dims)
            v16h aK;
            const float* kp = k + (sbase + n) * FEAT + h * HEAD_DIM;
            #pragma unroll
            for (int j = 0; j < 8; j++) {
                aK[j]     = (_Float16)kp[hi * 16 + j];
                aK[8 + j] = (_Float16)kp[hi * 16 + 8 + j];
            }
            v8f cS = {};
            cS = __builtin_amdgcn_wmma_f32_16x16x32_f16(false, aK, false, bQ,
                                                        (short)0, cS, false, false);
            #pragma unroll
            for (int r = 0; r < 8; r++) {
                const int   s  = sbase + hi * 8 + r;        // candidate index
                const float ys = (float)(s / W_);
                const float xs = (float)(s % W_);
                const float cc = mode ? (slp * xs + icp) : (slp * ys + icp);
                const float dd = mode ? (ys - cc) : (xs - cc);
                const bool  ok = (s != 0) && (dd < 2.0f) && (dd > -2.0f);
                const float sc = cS[r] * scale;
                msc[sub][r] = ok ? sc : -3.0e38f;
                if (ok) localmax = fmaxf(localmax, sc);
            }
        }

        // online softmax stats per pixel (lanes n and n+16 hold identical copies)
        localmax = fmaxf(localmax, __shfl_xor(localmax, 16));
        const float new_max = fmaxf(run_max, localmax);
        const float corr    = __expf(run_max - new_max);  // both -3e38 -> 1, harmless (sums are 0)

        // P values: the C-layout of the transposed-score WMMA IS the A-layout
        // (16 pixels x 32 candidates) of the PV WMMA after packing to f16.
        v16h aP;
        float psum = 0.0f;
        #pragma unroll
        for (int sub = 0; sub < 2; sub++)
            #pragma unroll
            for (int r = 0; r < 8; r++) {
                const float pe = (msc[sub][r] > -1.0e38f)
                                     ? __expf(msc[sub][r] - new_max) : 0.0f;
                psum += pe;
                aP[sub * 8 + r] = (_Float16)pe;
            }
        psum += __shfl_xor(psum, 16);
        run_sum = run_sum * corr + psum;
        run_max = new_max;

        // rescale accumulators: factor of row m = hi*8+r comes from lane m
        #pragma unroll
        for (int r = 0; r < 8; r++) {
            const float cm = __shfl(corr, hi * 8 + r);
            acc0[r] *= cm;
            acc1[r] *= cm;
        }

        // B fragments for PV: V chunk (32 candidates x 32 dims), two dim-halves
        v16h bV0, bV1;
        #pragma unroll
        for (int e = 0; e < 16; e++) {
            const float* vp = v + (chunk + (e >> 3) * 16 + hi * 8 + (e & 7)) * FEAT + h * HEAD_DIM + n;
            bV0[e] = (_Float16)vp[0];
            bV1[e] = (_Float16)vp[16];
        }
        acc0 = __builtin_amdgcn_wmma_f32_16x16x32_f16(false, aP, false, bV0,
                                                      (short)0, acc0, false, false);
        acc1 = __builtin_amdgcn_wmma_f32_16x16x32_f16(false, aP, false, bV1,
                                                      (short)0, acc1, false, false);
    }

    // ---- normalize and write: acc[r] = Out[row hi*8+r][dim n (+16)]
    #pragma unroll
    for (int r = 0; r < 8; r++) {
        const float sm  = __shfl(run_sum, hi * 8 + r);
        const float inv = (sm > 0.0f) ? (1.0f / sm) : 0.0f;  // all-masked row -> 0 (nan_to_num)
        const int   row = lt * 16 + hi * 8 + r;
        out[row * FEAT + h * HEAD_DIM + n]      = acc0[r] * inv;
        out[row * FEAT + h * HEAD_DIM + 16 + n] = acc1[r] * inv;
    }
}

extern "C" void kernel_launch(void* const* d_in, const int* in_sizes, int n_in,
                              void* d_out, int out_size, void* d_ws, size_t ws_size,
                              hipStream_t stream) {
    (void)in_sizes; (void)n_in; (void)out_size; (void)d_ws; (void)ws_size;
    const float* q  = (const float*)d_in[0];
    const float* k  = (const float*)d_in[1];
    const float* v  = (const float*)d_in[2];
    const float* K0 = (const float*)d_in[3];
    const float* K1 = (const float*)d_in[4];
    const float* R  = (const float*)d_in[5];
    const float* t  = (const float*)d_in[6];
    float* out = (float*)d_out;

    dim3 grid(L_ / 16, NHEAD);  // 80 pixel tiles x 8 heads = 640 single-wave WGs
    dim3 block(32);             // one wave32 per workgroup
    hipLaunchKernelGGL(One2ManyAttention_kernel, grid, block, 0, stream,
                       q, k, v, K0, K1, R, t, out);
}
